// MoEFSCIL_71545565216783
// MI455X (gfx1250) — matrix operand order repaired and checked by reading, not compiled, not run on hardware
//
#include <hip/hip_runtime.h>

// ---------------------------------------------------------------------------
// MoE FSCIL forward for MI455X (gfx1250).  All heavy GEMMs run through
// v_wmma_f32_16x16x32_f16; scans / softmax / LN are f32 scalar kernels.
// Round 2: n-major LDS B-tile so both A and B fragments are contiguous
// 2x ds_load_b128 reads (kills the v_mov_b16 packing storm of round 1).
// ---------------------------------------------------------------------------

typedef _Float16 v16h __attribute__((ext_vector_type(16)));
typedef float    v8f  __attribute__((ext_vector_type(8)));

struct alignas(16) HalfVec8 { _Float16 h[8]; };
union  Frag16 { v16h v; HalfVec8 h8[2]; };

struct Gate { int i0, i1; float w0, w1; };

#define TM 128
#define TN 128
#define TK 32

// ------------------------------ cast ---------------------------------------
__global__ __launch_bounds__(256) void cast_f32_f16(const float* __restrict__ in,
                                                    _Float16* __restrict__ out, long n) {
  for (long i = (long)blockIdx.x * 256 + threadIdx.x; i < n; i += (long)gridDim.x * 256)
    out[i] = (_Float16)in[i];
}

// ------------------------------ WMMA GEMM ----------------------------------
// Computes C[z] = A[z] @ Bw[z]^T + bias  (b_is_nk=1, Bw is [N,K] row-major)
//          C[z] = A[z] @ Bw[z]   + bias  (b_is_nk=0, Bw is [K,N] row-major)
// Two-level batch strides: off = (z/binner)*s_o + (z%binner)*s_i
__global__ __launch_bounds__(256) void gemm_f16_wmma(
    const _Float16* __restrict__ A, int lda, long sAo, long sAi,
    const _Float16* __restrict__ Bw, int ldb, long sBo, long sBi,
    void* __restrict__ Cp, int ldc, long sCo, long sCi,
    const float* __restrict__ bias,
    int M, int N, int K, int binner, int b_is_nk, int out_half, float alpha) {
  __shared__ _Float16 shA[TM][TK + 8];   // [m][k], row stride 80B (16B aligned)
  __shared__ _Float16 shB[TN][TK + 8];   // [n][k], row stride 80B (16B aligned)

  const int tid  = threadIdx.x;
  const int lane = tid & 31;
  const int wave = tid >> 5;
  const int z    = blockIdx.z;
  const long offA = (long)(z / binner) * sAo + (long)(z % binner) * sAi;
  const long offB = (long)(z / binner) * sBo + (long)(z % binner) * sBi;
  const long offC = (long)(z / binner) * sCo + (long)(z % binner) * sCi;
  const _Float16* Ab = A + offA;
  const _Float16* Bb = Bw + offB;

  const int mBase = blockIdx.y * TM;
  const int nBase = blockIdx.x * TN;
  const int wm0 = (wave & 3) * 32;   // 4 waves down
  const int wn0 = (wave >> 2) * 64;  // 2 waves across

  v8f acc[2][4];
#pragma unroll
  for (int i = 0; i < 2; ++i)
#pragma unroll
    for (int j = 0; j < 4; ++j) acc[i][j] = {};

  const int mfrag = lane & 15;       // row within 16x16 tile (A) / col (B,C)
  const int ksel  = lane >> 4;       // which K half-group this lane covers
  const int koff  = ksel * 16;

  for (int kt = 0; kt < K; kt += TK) {
    // ---- stage A tile [TM x TK] : straight row copies ----
    for (int s = tid; s < TM * 4; s += 256) {
      int r = s >> 2, c8 = (s & 3) * 8;
      HalfVec8 v{};
      if (mBase + r < M)
        v = *(const HalfVec8*)(Ab + (long)(mBase + r) * lda + kt + c8);
      *(HalfVec8*)(&shA[r][c8]) = v;
    }
    // ---- stage B tile as [TN x TK] (n-major) ----
    if (b_is_nk) {
      // W is [N,K]: row copies, no transpose needed.
      for (int s = tid; s < TN * 4; s += 256) {
        int n = s >> 2, c8 = (s & 3) * 8;
        HalfVec8 v{};
        if (nBase + n < N)
          v = *(const HalfVec8*)(Bb + (long)(nBase + n) * ldb + kt + c8);
        *(HalfVec8*)(&shB[n][c8]) = v;
      }
    } else {
      // B is [K,N]: transpose while staging.
      for (int s = tid; s < TK * 16; s += 256) {
        int r = s >> 4, c8 = (s & 15) * 8;
        HalfVec8 v{};
        if (nBase + c8 + 8 <= N) {
          v = *(const HalfVec8*)(Bb + (long)(kt + r) * ldb + nBase + c8);
        } else if (nBase + c8 < N) {
          for (int j = 0; j < 8 && nBase + c8 + j < N; ++j)
            v.h[j] = Bb[(long)(kt + r) * ldb + nBase + c8 + j];
        }
#pragma unroll
        for (int j = 0; j < 8; ++j) shB[c8 + j][r] = v.h[j];
      }
    }
    // prefetch next A tile rows while we compute
    if (kt + TK < K && (mBase + (tid >> 1)) < M)
      __builtin_prefetch(Ab + (long)(mBase + (tid >> 1)) * lda + kt + TK + (tid & 1) * 16, 0, 0);
    __syncthreads();

    // ---- fragments: each is two contiguous 16B LDS reads ----
    Frag16 af[2];
#pragma unroll
    for (int i = 0; i < 2; ++i) {
      int m = wm0 + i * 16 + mfrag;
      af[i].h8[0] = *(const HalfVec8*)(&shA[m][ksel * 8]);
      af[i].h8[1] = *(const HalfVec8*)(&shA[m][16 + ksel * 8]);
    }
    Frag16 bf[4];
#pragma unroll
    for (int j = 0; j < 4; ++j) {
      int n = wn0 + j * 16 + mfrag;
      bf[j].h8[0] = *(const HalfVec8*)(&shB[n][ksel * 8]);
      bf[j].h8[1] = *(const HalfVec8*)(&shB[n][16 + ksel * 8]);
    }
#pragma unroll
    for (int i = 0; i < 2; ++i)
#pragma unroll
      for (int j = 0; j < 4; ++j)
        acc[i][j] = __builtin_amdgcn_wmma_f32_16x16x32_f16(
            false, af[i].v, false, bf[j].v, (short)0, acc[i][j], false, false);
    __syncthreads();
  }

  // ---- epilogue ----
  const int mofs = ksel * 8;
#pragma unroll
  for (int i = 0; i < 2; ++i) {
#pragma unroll
    for (int j = 0; j < 4; ++j) {
      int n = nBase + wn0 + j * 16 + mfrag;
      if (n >= N) continue;
      float bv = bias ? bias[n] : 0.0f;
#pragma unroll
      for (int r = 0; r < 8; ++r) {
        int m = mBase + wm0 + i * 16 + mofs + r;
        if (m >= M) continue;
        float val = acc[i][j][r] * alpha + bv;
        long idx = offC + (long)m * ldc + n;
        if (out_half) ((_Float16*)Cp)[idx] = (_Float16)val;
        else          ((float*)Cp)[idx]    = val;
      }
    }
  }
}

// ------------------------------ softmax (256-wide rows) --------------------
__global__ __launch_bounds__(256) void softmax256(const float* __restrict__ S,
                                                  _Float16* __restrict__ P, float scale) {
  __shared__ float red[256];
  long row = blockIdx.x;
  int t = threadIdx.x;
  float v = S[row * 256 + t] * scale;
  red[t] = v; __syncthreads();
  for (int s = 128; s > 0; s >>= 1) { if (t < s) red[t] = fmaxf(red[t], red[t + s]); __syncthreads(); }
  float mx = red[0]; __syncthreads();
  float e = __expf(v - mx);
  red[t] = e; __syncthreads();
  for (int s = 128; s > 0; s >>= 1) { if (t < s) red[t] += red[t + s]; __syncthreads(); }
  P[row * 256 + t] = (_Float16)(e / red[0]);
}

// ------------------------------ k2 projection ------------------------------
__global__ __launch_bounds__(256) void k2_kernel(const float* __restrict__ eq,
                                                 const float* __restrict__ caw,
                                                 const float* __restrict__ cab,
                                                 float* __restrict__ k2) {
  int i = blockIdx.x * 256 + threadIdx.x;
  if (i >= 8 * 768) return;
  int e = i / 768, n = i % 768;
  const float* q = eq + e * 768;
  const float* w = caw + (long)(768 + n) * 768;
  float s = 0.f;
  for (int c = 0; c < 768; ++c) s += q[c] * w[c];
  k2[i] = s + cab[768 + n];
}

// ------------------------------ gating + top-2 -----------------------------
__global__ __launch_bounds__(256) void gating_kernel(const _Float16* __restrict__ q2,
                                                     const float* __restrict__ k2,
                                                     Gate* __restrict__ g) {
  __shared__ float sk2[8 * 768];
  __shared__ float red[256];
  __shared__ float accE[8];
  int b = blockIdx.x, t = threadIdx.x;
  for (int i = t; i < 8 * 768; i += 256) sk2[i] = k2[i];
  __syncthreads();
  const float scale = 0.10206207261596575f;  // 1/sqrt(96)
  float lacc[8] = {};
  for (int task = t; task < 256 * 8; task += 256) {
    int l = task >> 3, h = task & 7;
    const _Float16* q = q2 + ((long)b * 256 + l) * 768 + h * 96;
    float lg[8] = {};
    for (int d = 0; d < 96; ++d) {
      float qv = (float)q[d];
#pragma unroll
      for (int e = 0; e < 8; ++e) lg[e] += qv * sk2[e * 768 + h * 96 + d];
    }
    float mx = -1e30f;
#pragma unroll
    for (int e = 0; e < 8; ++e) mx = fmaxf(mx, lg[e] * scale);
    float sum = 0.f, ex[8];
#pragma unroll
    for (int e = 0; e < 8; ++e) { ex[e] = __expf(lg[e] * scale - mx); sum += ex[e]; }
#pragma unroll
    for (int e = 0; e < 8; ++e) lacc[e] += ex[e] / sum;
  }
  for (int e = 0; e < 8; ++e) {
    red[t] = lacc[e]; __syncthreads();
    for (int s = 128; s > 0; s >>= 1) { if (t < s) red[t] += red[t + s]; __syncthreads(); }
    if (t == 0) accE[e] = red[0] / (256.f * 8.f);
    __syncthreads();
  }
  if (t == 0) {
    float mx = -1e30f;
    for (int e = 0; e < 8; ++e) mx = fmaxf(mx, accE[e]);
    float s = 0.f, p[8];
    for (int e = 0; e < 8; ++e) { p[e] = __expf(accE[e] - mx); s += p[e]; }
    for (int e = 0; e < 8; ++e) p[e] /= s;
    int i0 = 0;
    for (int e = 1; e < 8; ++e) if (p[e] > p[i0]) i0 = e;
    int i1 = (i0 == 0) ? 1 : 0;
    for (int e = 0; e < 8; ++e) if (e != i0 && p[e] > p[i1]) i1 = e;
    float e0 = 1.0f, e1 = __expf(p[i1] - p[i0]);
    Gate gg; gg.i0 = i0; gg.i1 = i1; gg.w0 = e0 / (e0 + e1); gg.w1 = e1 / (e0 + e1);
    g[b] = gg;
  }
}

// ------------------------------ depthwise conv 3x3 + SiLU ------------------
__global__ __launch_bounds__(256) void dwconv_silu_kernel(const float* __restrict__ xz,
                                                          const float* __restrict__ cw,
                                                          const float* __restrict__ cb,
                                                          float* __restrict__ xc) {
  int p = blockIdx.x;           // (b, h, w)
  int b = p >> 8, hw = p & 255, h = hw >> 4, w = hw & 15;
  for (int c = threadIdx.x; c < 768; c += 256) {
    float s = 0.f;
    for (int dh = -1; dh <= 1; ++dh) {
      int hh = h + dh; if (hh < 0 || hh > 15) continue;
      for (int dw = -1; dw <= 1; ++dw) {
        int ww = w + dw; if (ww < 0 || ww > 15) continue;
        s += xz[(((long)b * 256 + hh * 16 + ww) * 1536) + c] * cw[c * 9 + (dh + 1) * 3 + (dw + 1)];
      }
    }
    s += cb[c];
    xc[(long)p * 768 + c] = s / (1.f + __expf(-s));
  }
}

// ------------------------------ x-projection (dbl, 4 dirs) -----------------
__global__ __launch_bounds__(256) void xproj_kernel(const float* __restrict__ xc,
                                                    const float* __restrict__ xw,  // [4][6][768]
                                                    float* __restrict__ dbl) {     // [4][16][256][6]
  int i = blockIdx.x * 256 + threadIdx.x;
  if (i >= 4 * 16 * 256 * 6) return;
  int j = i % 6, rest = i / 6;
  int p = rest % 256, b = (rest / 256) % 16, d = rest / (256 * 16);
  const float* x = xc + ((long)b * 256 + p) * 768;
  const float* w = xw + ((long)d * 6 + j) * 768;
  float s = 0.f;
  for (int c = 0; c < 768; ++c) s += x[c] * w[c];
  dbl[i] = s;
}

// ------------------------------ fused 4-direction selective scan -----------
__global__ __launch_bounds__(256) void scan_kernel(const float* __restrict__ xc,
                                                   const float* __restrict__ dbl,
                                                   const float* __restrict__ dtw,   // [4][768][4]
                                                   const float* __restrict__ dtb,   // [4][768]
                                                   const float* __restrict__ Alog,  // [4][768]
                                                   const float* __restrict__ Dp,    // [4][768]
                                                   float* __restrict__ ysum) {
  int b = blockIdx.x, t = threadIdx.x;
  for (int d = 0; d < 4; ++d) {
    float A[3], Dv[3], bd[3], w0[3], w1[3], w2[3], w3[3], hst[3];
#pragma unroll
    for (int cc = 0; cc < 3; ++cc) {
      int c = t + cc * 256;
      A[cc]  = -__expf(Alog[d * 768 + c]);
      Dv[cc] = Dp[d * 768 + c];
      bd[cc] = dtb[d * 768 + c];
      const float* w = dtw + ((long)d * 768 + c) * 4;
      w0[cc] = w[0]; w1[cc] = w[1]; w2[cc] = w[2]; w3[cc] = w[3];
      hst[cc] = 0.f;
    }
    for (int l = 0; l < 256; ++l) {
      int p;
      if (d == 0)      p = l;
      else if (d == 1) p = 255 - l;
      else if (d == 2) p = ((l & 15) << 4) | (l >> 4);
      else { int l2 = 255 - l; p = ((l2 & 15) << 4) | (l2 >> 4); }
      const float* db = dbl + (((long)(d * 16 + b) * 256 + p) * 6);
      float r0 = db[0], r1 = db[1], r2 = db[2], r3 = db[3], Bv = db[4], Cv = db[5];
#pragma unroll
      for (int cc = 0; cc < 3; ++cc) {
        int c = t + cc * 256;
        long xi = ((long)b * 256 + p) * 768 + c;
        float xv = xc[xi];
        float dtr = r0 * w0[cc] + r1 * w1[cc] + r2 * w2[cc] + r3 * w3[cc] + bd[cc];
        float dt = (dtr > 20.f) ? dtr : log1pf(__expf(dtr));
        hst[cc] = __expf(dt * A[cc]) * hst[cc] + dt * xv * Bv;
        float y = hst[cc] * Cv + Dv[cc] * xv;
        if (d == 0) ysum[xi] = y; else ysum[xi] += y;
      }
    }
  }
}

// ------------------------------ LN + SiLU gate + spatial mean --------------
__global__ __launch_bounds__(256) void ln_gate_mean_kernel(const float* __restrict__ ysum,
                                                           const float* __restrict__ xz,
                                                           const float* __restrict__ lns,
                                                           const float* __restrict__ lnb,
                                                           float* __restrict__ all_out, int e) {
  __shared__ float r1[256], r2[256];
  int b = blockIdx.x, t = threadIdx.x;
  float acc[3] = {0.f, 0.f, 0.f};
  for (int p = 0; p < 256; ++p) {
    const float* y = ysum + ((long)b * 256 + p) * 768;
    float v[3], s1 = 0.f, s2 = 0.f;
#pragma unroll
    for (int cc = 0; cc < 3; ++cc) { v[cc] = y[t + cc * 256]; s1 += v[cc]; s2 += v[cc] * v[cc]; }
    r1[t] = s1; r2[t] = s2; __syncthreads();
    for (int s = 128; s > 0; s >>= 1) { if (t < s) { r1[t] += r1[t + s]; r2[t] += r2[t + s]; } __syncthreads(); }
    float mu  = r1[0] / 768.f;
    float var = r2[0] / 768.f - mu * mu;
    float inv = rsqrtf(var + 1e-5f);
    __syncthreads();
#pragma unroll
    for (int cc = 0; cc < 3; ++cc) {
      int c = t + cc * 256;
      float yn = (v[cc] - mu) * inv * lns[c] + lnb[c];
      float z = xz[((long)b * 256 + p) * 1536 + 768 + c];
      acc[cc] += yn * (z / (1.f + __expf(-z)));
    }
  }
#pragma unroll
  for (int cc = 0; cc < 3; ++cc)
    all_out[((long)e * 16 + b) * 768 + t + cc * 256] = acc[cc] / 256.f;
}

// ------------------------------ top-2 combine ------------------------------
__global__ __launch_bounds__(256) void combine_kernel(const float* __restrict__ all_out,
                                                      const Gate* __restrict__ g,
                                                      float* __restrict__ out) {
  int b = blockIdx.x, t = threadIdx.x;
  Gate gg = g[b];
#pragma unroll
  for (int cc = 0; cc < 3; ++cc) {
    int c = t + cc * 256;
    out[(long)b * 768 + c] = gg.w0 * all_out[((long)gg.i0 * 16 + b) * 768 + c] +
                             gg.w1 * all_out[((long)gg.i1 * 16 + b) * 768 + c];
  }
}

// ------------------------------ host orchestration -------------------------
static void launch_gemm(hipStream_t stream,
                        const _Float16* A, int lda, long sAo, long sAi,
                        const _Float16* Bw, int ldb, long sBo, long sBi,
                        void* C, int ldc, long sCo, long sCi,
                        const float* bias, int M, int N, int K,
                        int binner, int b_is_nk, int out_half, float alpha, int nbatch) {
  dim3 grid((N + TN - 1) / TN, (M + TM - 1) / TM, nbatch);
  gemm_f16_wmma<<<grid, 256, 0, stream>>>(A, lda, sAo, sAi, Bw, ldb, sBo, sBi,
                                          C, ldc, sCo, sCi, bias, M, N, K,
                                          binner, b_is_nk, out_half, alpha);
}

extern "C" void kernel_launch(void* const* d_in, const int* in_sizes, int n_in,
                              void* d_out, int out_size, void* d_ws, size_t ws_size,
                              hipStream_t stream) {
  (void)in_sizes; (void)n_in; (void)out_size; (void)ws_size;
  const float* x        = (const float*)d_in[0];
  const float* sa_in_w  = (const float*)d_in[1];
  const float* sa_in_b  = (const float*)d_in[2];
  const float* sa_out_w = (const float*)d_in[3];
  const float* sa_out_b = (const float*)d_in[4];
  const float* ca_in_w  = (const float*)d_in[5];
  const float* ca_in_b  = (const float*)d_in[6];
  const float* eq       = (const float*)d_in[7];
  const float* e_in_w   = (const float*)d_in[8];
  const float* e_in_b   = (const float*)d_in[9];
  const float* e_conv_w = (const float*)d_in[10];
  const float* e_conv_b = (const float*)d_in[11];
  const float* e_xproj  = (const float*)d_in[12];
  const float* e_dtp_w  = (const float*)d_in[13];
  const float* e_dtp_b  = (const float*)d_in[14];
  const float* e_Alog   = (const float*)d_in[15];
  const float* e_D      = (const float*)d_in[16];
  const float* e_ln_s   = (const float*)d_in[17];
  const float* e_ln_b   = (const float*)d_in[18];

  char* ws = (char*)d_ws;
  size_t cur = 0;
  auto alloc = [&](size_t bytes) -> void* {
    void* p = ws + cur;
    cur = (cur + bytes + 255) & ~((size_t)255);
    return p;
  };

  _Float16* xs_h   = (_Float16*)alloc(4096L * 768 * 2);
  _Float16* wqkv_h = (_Float16*)alloc(2304L * 768 * 2);
  _Float16* wout_h = (_Float16*)alloc(768L * 768 * 2);
  _Float16* wca_h  = (_Float16*)alloc(768L * 768 * 2);
  _Float16* we_h   = (_Float16*)alloc(8L * 1536 * 768 * 2);
  _Float16* qkv_h  = (_Float16*)alloc(4096L * 2304 * 2);
  float*    scores = (float*)   alloc(128L * 256 * 256 * 4);
  _Float16* P_h    = (_Float16*)alloc(128L * 256 * 256 * 2);
  _Float16* ctx_h  = (_Float16*)alloc(4096L * 768 * 2);
  _Float16* ctx2_h = (_Float16*)alloc(4096L * 768 * 2);
  _Float16* q2_h   = (_Float16*)alloc(4096L * 768 * 2);
  float*    k2_f   = (float*)   alloc(8L * 768 * 4);
  Gate*     gate   = (Gate*)    alloc(16 * sizeof(Gate));
  float*    xz_f   = (float*)   alloc(4096L * 1536 * 4);
  float*    xc_f   = (float*)   alloc(4096L * 768 * 4);
  float*    dbl_f  = (float*)   alloc(4L * 16 * 256 * 6 * 4);
  float*    ysum_f = (float*)   alloc(4096L * 768 * 4);
  float*    allout = (float*)   alloc(8L * 16 * 768 * 4);

  // ---- casts to f16 ----
  cast_f32_f16<<<1024, 256, 0, stream>>>(x,        xs_h,   4096L * 768);
  cast_f32_f16<<<1024, 256, 0, stream>>>(sa_in_w,  wqkv_h, 2304L * 768);
  cast_f32_f16<<<512,  256, 0, stream>>>(sa_out_w, wout_h, 768L * 768);
  cast_f32_f16<<<512,  256, 0, stream>>>(ca_in_w,  wca_h,  768L * 768);      // rows [0,768)
  cast_f32_f16<<<2048, 256, 0, stream>>>(e_in_w,   we_h,   8L * 1536 * 768);

  // ---- qkv = xs @ sa_in_w^T + b  (f16 out) ----
  launch_gemm(stream, xs_h, 768, 0, 0, wqkv_h, 768, 0, 0,
              qkv_h, 2304, 0, 0, sa_in_b, 4096, 2304, 768, 1, 1, 1, 1.f, 1);

  // ---- scores[z=(b,h)] = Q @ K^T  (f32 out) ----
  launch_gemm(stream, qkv_h, 2304, 256L * 2304, 96, qkv_h + 768, 2304, 256L * 2304, 96,
              scores, 256, 8L * 256 * 256, 256L * 256, nullptr,
              256, 256, 96, 8, 1, 0, 1.f, 128);
  softmax256<<<32768, 256, 0, stream>>>(scores, P_h, 0.10206207261596575f);

  // ---- ctx[z] = P @ V  (f16 out, written into [B,L,DIM] layout) ----
  launch_gemm(stream, P_h, 256, 8L * 256 * 256, 256L * 256, qkv_h + 1536, 2304, 256L * 2304, 96,
              ctx_h, 768, 256L * 768, 96, nullptr,
              256, 96, 256, 8, 0, 1, 1.f, 128);

  // ---- sa_out projection and cross-attn query projection ----
  launch_gemm(stream, ctx_h, 768, 0, 0, wout_h, 768, 0, 0,
              ctx2_h, 768, 0, 0, sa_out_b, 4096, 768, 768, 1, 1, 1, 1.f, 1);
  launch_gemm(stream, ctx2_h, 768, 0, 0, wca_h, 768, 0, 0,
              q2_h, 768, 0, 0, ca_in_b, 4096, 768, 768, 1, 1, 1, 1.f, 1);

  // ---- expert-key projection + gating/top-2 ----
  k2_kernel<<<24, 256, 0, stream>>>(eq, ca_in_w, ca_in_b, k2_f);
  gating_kernel<<<16, 256, 0, stream>>>(q2_h, k2_f, gate);

  // ---- experts ----
  for (int e = 0; e < 8; ++e) {
    launch_gemm(stream, xs_h, 768, 0, 0, we_h + (long)e * 1536 * 768, 768, 0, 0,
                xz_f, 1536, 0, 0, e_in_b + (long)e * 1536, 4096, 1536, 768, 1, 1, 0, 1.f, 1);
    dwconv_silu_kernel<<<4096, 256, 0, stream>>>(xz_f, e_conv_w + (long)e * 768 * 9,
                                                 e_conv_b + (long)e * 768, xc_f);
    xproj_kernel<<<(4 * 16 * 256 * 6 + 255) / 256, 256, 0, stream>>>(
        xc_f, e_xproj + (long)e * 4 * 6 * 768, dbl_f);
    scan_kernel<<<16, 256, 0, stream>>>(xc_f, dbl_f,
                                        e_dtp_w + (long)e * 4 * 768 * 4,
                                        e_dtp_b + (long)e * 4 * 768,
                                        e_Alog  + (long)e * 4 * 768,
                                        e_D     + (long)e * 4 * 768, ysum_f);
    ln_gate_mean_kernel<<<16, 256, 0, stream>>>(ysum_f, xz_f,
                                                e_ln_s + (long)e * 768,
                                                e_ln_b + (long)e * 768, allout, e);
  }

  combine_kernel<<<16, 256, 0, stream>>>(allout, gate, (float*)d_out);
}
